// VarianceAdaptor_21406117003464
// MI455X (gfx1250) — compile-verified, matrix operand order checked by reading, not run on hardware
//
#include <hip/hip_runtime.h>

// ---------------------------------------------------------------------------
// FastSpeech2 VarianceAdaptor for MI455X (gfx1250, wave32, WMMA)
// B=32, T=512, C=CF=384, K=3, M=2048, NBINS=256
// Convs are GEMMs (Kdim = 3*384 = 1152) via v_wmma_f32_16x16x32_bf16.
// ---------------------------------------------------------------------------

#define NB    32
#define TT    512
#define CC    384
#define MM    2048
#define KSTEPS 36           // 1152/32
#define CTILES 24           // 384/16
#define CTSTRIDE 18432      // halves between ctiles in packed W: 36*512
#define KSSTRIDE 512        // halves between ksteps in packed W: 32*16

typedef __attribute__((ext_vector_type(16))) __bf16 v16bf;
typedef __attribute__((ext_vector_type(8)))  float  v8f;

__device__ __forceinline__ unsigned short f2bf(float f) {
  unsigned u = __builtin_bit_cast(unsigned, f);
  unsigned r = u + 0x7FFFu + ((u >> 16) & 1u);   // round to nearest even
  return (unsigned short)(r >> 16);
}
__device__ __forceinline__ float bf2f(unsigned short h) {
  unsigned u = ((unsigned)h) << 16;
  return __builtin_bit_cast(float, u);
}
__device__ __forceinline__ unsigned pk2(float a, float b) {
  return (unsigned)f2bf(a) | ((unsigned)f2bf(b) << 16);
}

// ---------------------------------------------------------------------------
// Pack conv weight (3,384,384) f32  ->  bf16 B-fragment layout:
// [ctile(24)][kstep(36)][lane(32)][slot(16)]
// lane = Nlocal + 16*((Klocal>>3)&1) ; slot = (Klocal&7) + 8*(Klocal>>4)
// ---------------------------------------------------------------------------
__global__ __launch_bounds__(256) void pack_w_kernel(
    const float* __restrict__ w, unsigned short* __restrict__ out) {
  int o = blockIdx.x * 256 + threadIdx.x;      // 24*36*32*16 = 442368 total
  int slot = o & 15;
  int lane = (o >> 4) & 31;
  int ks   = (o >> 9) % KSTEPS;
  int ct   = (o >> 9) / KSTEPS;
  int hi = lane >> 4;
  int nl = lane & 15;
  int kl = (slot & 7) + 8 * hi + 16 * (slot >> 3);   // Klocal in [0,32)
  int K  = ks * 32 + kl;                              // global Kdim index
  int cout = ct * 16 + nl;
  out[o] = f2bf(w[(size_t)K * CC + cout]);            // w[(tap*384+cin)*384+cout]
}

// ---------------------------------------------------------------------------
// Fused conv1d(K=3 SAME) + bias + ReLU + LayerNorm.
// Workgroup (128 thr = 4 waves) -> 32 rows x 384 cols.
// Wave w owns col tiles [w*6, w*6+6) against 2 row fragments: 12 WMMA/k-step
// sharing 6 B fragments. B loads are issued BEFORE the staging barrier
// (they don't depend on it, and the barrier is a scheduling-region boundary,
// so the scheduler cannot sink them back to just-in-time position) ->
// barrier + A ds_loads overlap B load latency, WMMAs drain with staggered
// waits. Ping-pong A staging: 1 barrier per k-step.
// Input row-major [rows][384] f32 (IN_BF16=false) or bf16 (true); output bf16.
// ---------------------------------------------------------------------------
template <bool IN_BF16>
__global__ __launch_bounds__(128) void conv_ln_kernel(
    const void* __restrict__ in_v, const unsigned short* __restrict__ Wp,
    const float* __restrict__ bias, const float* __restrict__ gamma,
    const float* __restrict__ beta, unsigned short* __restrict__ out, int L) {
  __shared__ __bf16 Afrag[2][2 * 32 * 16];   // [buf][frag*512 + lane*16 + slot], 4KB
  __shared__ float  hrow[32 * CC];           // conv result tile, 48 KB

  const int tid  = threadIdx.x;
  const int wave = tid >> 5;
  const int lane = tid & 31;
  const int tilesPerBatch = L >> 5;
  const int batch = blockIdx.x / tilesPerBatch;
  const int r0    = (blockIdx.x % tilesPerBatch) << 5;

  const float*          inF = (const float*)in_v;
  const unsigned short* inH = (const unsigned short*)in_v;
  const size_t inRowBase = (size_t)batch * L;

  v8f acc[2][6];
#pragma unroll
  for (int f = 0; f < 2; ++f)
#pragma unroll
    for (int t = 0; t < 6; ++t)
      acc[f][t] = (v8f){0.f,0.f,0.f,0.f,0.f,0.f,0.f,0.f};

  // A staging map: thread -> (row Mst, 8-wide K group kb). Each aligned 8-K
  // group is one contiguous (lane, slot-run) in the fragment -> 1 b128 store.
  const int Mst = tid >> 2;             // 0..31
  const int kb  = (tid & 3) * 8;        // 0,8,16,24
  const int fst = Mst >> 4;             // row fragment 0/1
  const int stHalf = fst * 512 + ((Mst & 15) + 16 * ((kb >> 3) & 1)) * 16
                   + ((kb >> 4) * 8);   // multiple of 8 halves (16B aligned)

  // B fragments: one base pointer per (wave,lane); all loads use immediate
  // offsets t*36864B, advancing 1024B per k-step.
  const __bf16* bp = (const __bf16*)Wp + (size_t)wave * 6 * CTSTRIDE + lane * 16;

  for (int ks = 0; ks < KSTEPS; ++ks) {
    const int tap   = ks / 12;               // 384/32 = 12 ksteps per tap
    const int cbase = (ks - tap * 12) * 32 + kb;
    const int rloc  = r0 + Mst + tap - 1;    // SAME conv halo
    uint4 pk;
    if (rloc < 0 || rloc >= L) {
      pk = (uint4){0u, 0u, 0u, 0u};
    } else {
      const size_t base = (inRowBase + rloc) * CC + cbase;
      if (IN_BF16) {
        pk = *(const uint4*)(inH + base);
      } else {
        const float4 f0 = *(const float4*)(inF + base);
        const float4 f1 = *(const float4*)(inF + base + 4);
        pk.x = pk2(f0.x, f0.y); pk.y = pk2(f0.z, f0.w);
        pk.z = pk2(f1.x, f1.y); pk.w = pk2(f1.z, f1.w);
      }
    }
    __bf16* Ab = Afrag[ks & 1];
    *(uint4*)(Ab + stHalf) = pk;             // ds_store_b128

    // Issue all 6 B-fragment loads BEFORE the barrier: their latency is
    // covered by the barrier wait + A fragment ds_loads.
    v16bf bfrag[6];
#pragma unroll
    for (int t = 0; t < 6; ++t)
      bfrag[t] = *(const v16bf*)(bp + (size_t)t * CTSTRIDE);  // imm offsets
    if (ks + 1 < KSTEPS)
      __builtin_prefetch(bp + KSSTRIDE, 0, 3);

    __syncthreads();

    const v16bf a0 = *(const v16bf*)(Ab + lane * 16);
    const v16bf a1 = *(const v16bf*)(Ab + 512 + lane * 16);
#pragma unroll
    for (int t = 0; t < 6; ++t) {
      acc[0][t] = __builtin_amdgcn_wmma_f32_16x16x32_bf16(
          false, a0, false, bfrag[t], (short)0, acc[0][t], false, false);
      acc[1][t] = __builtin_amdgcn_wmma_f32_16x16x32_bf16(
          false, a1, false, bfrag[t], (short)0, acc[1][t], false, false);
    }
    bp += KSSTRIDE;
  }

  __syncthreads();
  // C/D layout: VGPR v -> row v (lanes 0-15) or v+8 (lanes 16-31); col = lane&15
  const int rowHalf = (lane >> 4) * 8;
  const int nl      = lane & 15;
#pragma unroll
  for (int f = 0; f < 2; ++f) {
#pragma unroll
    for (int t = 0; t < 6; ++t) {
      const int col = wave * 96 + t * 16 + nl;
      const float bcol = bias[col];
#pragma unroll
      for (int v = 0; v < 8; ++v) {
        float val = acc[f][t][v] + bcol;
        val = val > 0.f ? val : 0.f;               // ReLU
        hrow[(f * 16 + rowHalf + v) * CC + col] = val;
      }
    }
  }
  __syncthreads();

  // LayerNorm over 384 cols: 4 threads per row, shuffle reduce within quad
  const int row = tid >> 2;
  const int sub = tid & 3;
  float s = 0.f, ss = 0.f;
  for (int c = sub; c < CC; c += 4) {
    const float v = hrow[row * CC + c];
    s += v; ss += v * v;
  }
#pragma unroll
  for (int off = 2; off >= 1; off >>= 1) {
    s  += __shfl_xor(s,  off, 32);
    ss += __shfl_xor(ss, off, 32);
  }
  const float mu  = s * (1.f / CC);
  const float var = ss * (1.f / CC) - mu * mu;
  const float rs  = rsqrtf(var + 1e-5f);
  const size_t orow = (inRowBase + r0 + row) * CC;
  for (int c = sub; c < CC; c += 4) {
    const float v = (hrow[row * CC + c] - mu) * rs * gamma[c] + beta[c];
    out[orow + c] = f2bf(v);
  }
}

// ---------------------------------------------------------------------------
// pred[row] = mask[row] ? dot(h[row,:], wl) + bl : 0   (one wave per row)
// ---------------------------------------------------------------------------
__global__ __launch_bounds__(256) void linear_mask_kernel(
    const unsigned short* __restrict__ h, const float* __restrict__ wl,
    const float* __restrict__ bl, const int* __restrict__ mask,
    float* __restrict__ out, int rows) {
  const int row  = blockIdx.x * 8 + (threadIdx.x >> 5);
  const int lane = threadIdx.x & 31;
  if (row >= rows) return;
  const unsigned short* hp = h + (size_t)row * CC;
  float s = 0.f;
  for (int c = lane; c < CC; c += 32) s += bf2f(hp[c]) * wl[c];
#pragma unroll
  for (int off = 16; off >= 1; off >>= 1) s += __shfl_xor(s, off, 32);
  if (lane == 0) out[row] = (mask[row] == 0) ? 0.f : (s + bl[0]);
}

// ---------------------------------------------------------------------------
// Per-batch serial cumsum of durations (32 batches, trivial cost)
// ---------------------------------------------------------------------------
__global__ void cumsum_kernel(const int* __restrict__ dur, int* __restrict__ cum,
                              int* __restrict__ totals, float* __restrict__ o_mellen) {
  const int b = threadIdx.x;
  if (b >= NB) return;
  int s = 0;
  for (int t = 0; t < TT; ++t) { s += dur[b * TT + t]; cum[b * TT + t] = s; }
  totals[b]   = s;
  o_mellen[b] = (float)s;
}

// ---------------------------------------------------------------------------
// Length regulate: xe[b,m,:] = x[b, min(searchsorted_right(cum[b],m),T-1), :]
// zeroed where m >= total. One block per (b,m).
// ---------------------------------------------------------------------------
__global__ __launch_bounds__(128) void expand_kernel(
    const float* __restrict__ x, const int* __restrict__ cum,
    const int* __restrict__ totals, float* __restrict__ xe) {
  const int bm = blockIdx.x;
  const int b  = bm >> 11;        // M = 2048
  const int m  = bm & (MM - 1);
  const int* c = cum + b * TT;
  const bool valid = m < totals[b];
  int lo = 0, hi = TT;            // first idx with c[idx] > m
  while (lo < hi) { const int mid = (lo + hi) >> 1; if (c[mid] <= m) lo = mid + 1; else hi = mid; }
  const int idx = lo < (TT - 1) ? lo : (TT - 1);
  const float* src = x  + ((size_t)b * TT + idx) * CC;
  float*       dst = xe + ((size_t)b * MM + m)   * CC;
  for (int k = threadIdx.x; k < CC; k += 128) dst[k] = valid ? src[k] : 0.f;
}

// ---------------------------------------------------------------------------
// xe[row,:] += emb[searchsorted_left(bins255, target[row]), :]
// ---------------------------------------------------------------------------
__global__ __launch_bounds__(128) void emb_add_kernel(
    float* __restrict__ xe, const float* __restrict__ target,
    const float* __restrict__ bins, const float* __restrict__ emb) {
  const int row = blockIdx.x;
  const float t = target[row];
  int lo = 0, hi = 255;           // first i with bins[i] >= t
  while (lo < hi) { const int mid = (lo + hi) >> 1; if (bins[mid] < t) lo = mid + 1; else hi = mid; }
  const float* e = emb + (size_t)lo * CC;
  float* dst = xe + (size_t)row * CC;
  for (int k = threadIdx.x; k < CC; k += 128) dst[k] += e[k];
}

__global__ void mask_copy_kernel(const int* __restrict__ m, float* __restrict__ o, int n) {
  const int i = blockIdx.x * 256 + threadIdx.x;
  if (i < n) o[i] = (float)m[i];
}

// ---------------------------------------------------------------------------
extern "C" void kernel_launch(void* const* d_in, const int* in_sizes, int n_in,
                              void* d_out, int out_size, void* d_ws, size_t ws_size,
                              hipStream_t stream) {
  (void)in_sizes; (void)n_in; (void)out_size; (void)ws_size;

  // setup_inputs() order (params dict flattened in insertion order)
  const float* x          = (const float*)d_in[0];
  const int*   src_mask   = (const int*)d_in[1];
  const int*   mel_mask   = (const int*)d_in[2];
  const float* pitch_t    = (const float*)d_in[3];
  const float* energy_t   = (const float*)d_in[4];
  const int*   dur_t      = (const int*)d_in[5];
  // d_in[6] = max_mel_len (scalar, fixed at 2048 here)
  struct VP { const float *w1,*b1,*g1,*be1,*w2,*b2,*g2,*be2,*wl,*bl; };
  auto getvp = [&](int base) {
    VP p;
    p.w1 = (const float*)d_in[base + 0]; p.b1 = (const float*)d_in[base + 1];
    p.g1 = (const float*)d_in[base + 2]; p.be1= (const float*)d_in[base + 3];
    p.w2 = (const float*)d_in[base + 4]; p.b2 = (const float*)d_in[base + 5];
    p.g2 = (const float*)d_in[base + 6]; p.be2= (const float*)d_in[base + 7];
    p.wl = (const float*)d_in[base + 8]; p.bl = (const float*)d_in[base + 9];
    return p;
  };
  const VP dur = getvp(7), pit = getvp(17), ene = getvp(27);
  const float* pitch_emb   = (const float*)d_in[37];
  const float* energy_emb  = (const float*)d_in[38];
  const float* pitch_bins  = (const float*)d_in[39];
  const float* energy_bins = (const float*)d_in[40];

  // Output layout (tuple concatenated flat, all float32)
  float* o_xe      = (float*)d_out;                       // 32*2048*384
  float* o_logdur  = o_xe     + (size_t)NB * MM * CC;     // 32*512
  float* o_pitch   = o_logdur + (size_t)NB * TT;          // 32*2048
  float* o_energy  = o_pitch  + (size_t)NB * MM;          // 32*2048
  float* o_mellen  = o_energy + (size_t)NB * MM;          // 32
  float* o_melmask = o_mellen + NB;                       // 32*2048

  // Workspace layout
  const size_t PW = (size_t)CTILES * KSTEPS * 32 * 16;    // halves per packed weight
  char* ws = (char*)d_ws;
  unsigned short* pw[6];
  for (int i = 0; i < 6; ++i) pw[i] = (unsigned short*)(ws + (size_t)i * PW * 2);
  unsigned short* h1 = (unsigned short*)(ws + 6 * PW * 2);
  unsigned short* h2 = h1 + (size_t)NB * MM * CC;
  int* cum    = (int*)(h2 + (size_t)NB * MM * CC);
  int* totals = cum + NB * TT;

  const int packBlocks = (CTILES * KSTEPS * 32 * 16) / 256;   // 1728
  pack_w_kernel<<<packBlocks, 256, 0, stream>>>(dur.w1, pw[0]);
  pack_w_kernel<<<packBlocks, 256, 0, stream>>>(dur.w2, pw[1]);
  pack_w_kernel<<<packBlocks, 256, 0, stream>>>(pit.w1, pw[2]);
  pack_w_kernel<<<packBlocks, 256, 0, stream>>>(pit.w2, pw[3]);
  pack_w_kernel<<<packBlocks, 256, 0, stream>>>(ene.w1, pw[4]);
  pack_w_kernel<<<packBlocks, 256, 0, stream>>>(ene.w2, pw[5]);

  // --- duration predictor on x: rows = 32*512, L = 512 ---
  const int rowsT = NB * TT, tilesT = rowsT / 32;
  conv_ln_kernel<false><<<tilesT, 128, 0, stream>>>(x,  pw[0], dur.b1, dur.g1, dur.be1, h1, TT);
  conv_ln_kernel<true ><<<tilesT, 128, 0, stream>>>(h1, pw[1], dur.b2, dur.g2, dur.be2, h2, TT);
  linear_mask_kernel<<<rowsT / 8, 256, 0, stream>>>(h2, dur.wl, dur.bl, src_mask, o_logdur, rowsT);

  // --- length regulate ---
  cumsum_kernel<<<1, 32, 0, stream>>>(dur_t, cum, totals, o_mellen);
  expand_kernel<<<NB * MM, 128, 0, stream>>>(x, cum, totals, o_xe);

  // --- pitch predictor on xe: rows = 32*2048, L = 2048 ---
  const int rowsM = NB * MM, tilesM = rowsM / 32;
  conv_ln_kernel<false><<<tilesM, 128, 0, stream>>>(o_xe, pw[2], pit.b1, pit.g1, pit.be1, h1, MM);
  conv_ln_kernel<true ><<<tilesM, 128, 0, stream>>>(h1,   pw[3], pit.b2, pit.g2, pit.be2, h2, MM);
  linear_mask_kernel<<<rowsM / 8, 256, 0, stream>>>(h2, pit.wl, pit.bl, mel_mask, o_pitch, rowsM);

  // xe += pitch_emb[bucketize(pitch_target)]
  emb_add_kernel<<<rowsM, 128, 0, stream>>>(o_xe, pitch_t, pitch_bins, pitch_emb);

  // --- energy predictor on updated xe ---
  conv_ln_kernel<false><<<tilesM, 128, 0, stream>>>(o_xe, pw[4], ene.b1, ene.g1, ene.be1, h1, MM);
  conv_ln_kernel<true ><<<tilesM, 128, 0, stream>>>(h1,   pw[5], ene.b2, ene.g2, ene.be2, h2, MM);
  linear_mask_kernel<<<rowsM / 8, 256, 0, stream>>>(h2, ene.wl, ene.bl, mel_mask, o_energy, rowsM);

  // xe += energy_emb[bucketize(energy_target)]
  emb_add_kernel<<<rowsM, 128, 0, stream>>>(o_xe, energy_t, energy_bins, energy_emb);

  // mel_mask passthrough (as float)
  mask_copy_kernel<<<(NB * MM) / 256, 256, 0, stream>>>(mel_mask, o_melmask, NB * MM);
}
